// Network_80771154969240
// MI455X (gfx1250) — matrix operand order repaired and checked by reading, not compiled, not run on hardware
//
#include <hip/hip_runtime.h>
#include <hip/hip_bf16.h>

// ---------------------------------------------------------------------------
// Recurrent rate network with Markram STP, MI455X (gfx1250, wave32) version.
//   - weights converted once f32 -> bf16 and packed into WMMA B-fragment layout
//   - per step: stp elementwise kernel + fused (WMMA GEMM + epilogue) kernel
//   - v_wmma_f32_16x16x32_bf16, M=16 == batch, f32 accumulation
//   - two independent accumulator chains per wave to break WMMA->WMMA RAW
// ---------------------------------------------------------------------------

namespace {
constexpr int   B_      = 16;
constexpr int   T_      = 20;
constexpr int   N_      = 4096;
constexpr int   NE_     = 3072;
constexpr int   NT_AB   = N_  / 16;   // 256 n-tiles (full matrix)
constexpr int   NT_STP  = NE_ / 16;   // 192 n-tiles (E block)
constexpr int   KT_AB   = N_  / 32;   // 128 k-tiles
constexpr int   KT_STP  = NE_ / 32;   // 96  k-tiles
constexpr float DT      = 0.05f;
constexpr float USE_    = 0.03f;
constexpr float INV_TAU_FAC = 1.0f / 1.0f;
constexpr float INV_TAU_REC = 1.0f / 0.25f;
constexpr float E_SYN   = 0.9f;   // EXP_DT_TAU_SYN
constexpr float E_TAU   = 0.8f;   // EXP_DT_TAU
}

typedef __bf16 bf16_t;
typedef bf16_t v16bf __attribute__((ext_vector_type(16)));
typedef bf16_t v8bf  __attribute__((ext_vector_type(8)));
typedef float  v8f   __attribute__((ext_vector_type(8)));

// ---------------------------------------------------------------------------
// One-time: convert f32 weight matrix [K x Ncols] (row-major, element (k,n) at
// k*Ncols+n) into bf16 packed B-fragment tiles.  Tile (kt, nt) is 512 bf16:
//   packed[((nt*KT + kt)*32 + lane)*16 + j]
//     lane <  16 : element (K = kt*32 + j      , N = nt*16 + lane)
//     lane >= 16 : element (K = kt*32 + 16 + j , N = nt*16 + lane-16)
// in contiguous K order, the same one load_a_frag uses (lanes 0-15: K=0..15,
// lanes 16-31: K=16..31; 2 K-values per VGPR, 8 VGPRs per lane).
// ---------------------------------------------------------------------------
__global__ __launch_bounds__(256) void pack_w_bf16(
    const float* __restrict__ src, __hip_bfloat16* __restrict__ dst,
    int K, int Ncols)
{
    long long tid   = (long long)blockIdx.x * blockDim.x + threadIdx.x;
    long long total = (long long)K * Ncols;
    if (tid >= total) return;
    int n    = (int)(tid % Ncols);           // coalesced read over n
    int k    = (int)(tid / Ncols);
    int kt   = k >> 5;
    int koff = k & 31;
    int nt   = n >> 4;
    int lane = (n & 15) | (koff & 16);
    int j    = koff & 15;
    int KT   = K >> 5;
    long long d = (((long long)nt * KT + kt) * 32 + lane) * 16 + j;
    dst[d] = __float2bfloat16(src[tid]);
}

// ---------------------------------------------------------------------------
// rates0 = relu(ff[:,0] + rec0 - thresh);  rec = rec0;  u = USE;  x = 1
// also emit rates in bf16 (A operand for step 0's GEMM).
// ---------------------------------------------------------------------------
__global__ __launch_bounds__(256) void init_state(
    const float* __restrict__ ff, const float* __restrict__ rec0,
    const float* __restrict__ thresh,
    float* __restrict__ rates, float* __restrict__ rec,
    float* __restrict__ u, float* __restrict__ x,
    __hip_bfloat16* __restrict__ rates_bf)
{
    int tid = blockIdx.x * blockDim.x + threadIdx.x;   // over B*N
    if (tid >= B_ * N_) return;
    int b = tid / N_;
    int n = tid - b * N_;
    float r0 = ff[((long long)b * T_ + 0) * N_ + n] + rec0[tid] - thresh[tid];
    r0 = fmaxf(r0, 0.0f);
    rates[tid]    = r0;
    rates_bf[tid] = __float2bfloat16(r0);
    rec[tid]      = rec0[tid];
    if (n < NE_) {
        int e = b * NE_ + n;
        u[e] = USE_;
        x[e] = 1.0f;
    }
}

// ---------------------------------------------------------------------------
// Per-step STP update (E block):  u,x RMW in place, Aux -> bf16 A operand.
// ---------------------------------------------------------------------------
__global__ __launch_bounds__(256) void stp_step(
    const float* __restrict__ rates,
    float* __restrict__ u, float* __restrict__ x,
    __hip_bfloat16* __restrict__ aux_bf)
{
    int tid = blockIdx.x * blockDim.x + threadIdx.x;   // over B*NE
    if (tid >= B_ * NE_) return;
    int b = tid / NE_;
    int i = tid - b * NE_;
    float rE = rates[b * N_ + i];
    float uu = u[tid];
    float xx = x[tid];
    float un = uu + DT * ((USE_ - uu) * INV_TAU_FAC + USE_ * (1.0f - uu) * rE);
    float xn = xx + DT * ((1.0f - xx) * INV_TAU_REC - un * xx * rE);
    u[tid] = un;
    x[tid] = xn;
    aux_bf[tid] = __float2bfloat16(un * xn * rE);     // J_STP == 1.0
}

// A-fragment loader, contiguous K order (matches packed B).  Lane (l&15) = row M; lanes 0-15
// hold K = 0..15, lanes 16-31 hold K = 16..31.  Row data is
// contiguous bf16, so this is two 16-byte loads.
__device__ inline v16bf load_a_frag(const bf16_t* __restrict__ row_kbase, int hi)
{
    const bf16_t* p = row_kbase + (hi ? 16 : 0);
    v8bf lo = *(const v8bf*)(p);
    v8bf hh = *(const v8bf*)(p + 8);
    return __builtin_shufflevector(lo, hh,
        0, 1, 2, 3, 4, 5, 6, 7, 8, 9, 10, 11, 12, 13, 14, 15);
}

// ---------------------------------------------------------------------------
// One recurrence step:  hidden-tile GEMM (Wab + STP correction) with
// v_wmma_f32_16x16x32_bf16, 4-wave K-split + LDS reduction, fused epilogue:
//   rec' = rec*0.9 + hidden*0.1 ; net = ff_t + rec'
//   rates' = rates*0.8 + relu(net - thresh)*0.2 ; out[:,t,:] = rates'
// One block = one 16-column n-tile; 256 blocks cover N=4096.
// Each wave runs TWO independent WMMA accumulation chains (even/odd k-tile)
// so consecutive WMMAs never share a C operand (no RAW stall on the XDL pipe).
// ---------------------------------------------------------------------------
__global__ __launch_bounds__(128, 1) void gemm_step(
    const __hip_bfloat16* __restrict__ ratesA,   // [16,4096] bf16 (A, read buf)
    const __hip_bfloat16* __restrict__ auxA,     // [16,3072] bf16
    const __hip_bfloat16* __restrict__ wabP,     // packed Wab_T
    const __hip_bfloat16* __restrict__ wstpP,    // packed W_stp_T
    const float* __restrict__ ff,                // [B,T,N]
    const float* __restrict__ thresh,            // [B,N]
    float* __restrict__ rates,                   // f32 carry (RMW own tile)
    float* __restrict__ rec,                     // f32 carry (RMW own tile)
    __hip_bfloat16* __restrict__ ratesB_next,    // bf16 A for step t+1
    float* __restrict__ out,                     // [B,T,N]
    int t)
{
    const int nt   = blockIdx.x;          // 0..255
    const int wave = threadIdx.x >> 5;    // 0..3 : K split
    const int lane = threadIdx.x & 31;
    const int mrow = lane & 15;           // A row (== batch within half)
    const int hi   = lane >> 4;

    v8f acc0 = {0.f, 0.f, 0.f, 0.f, 0.f, 0.f, 0.f, 0.f};
    v8f acc1 = {0.f, 0.f, 0.f, 0.f, 0.f, 0.f, 0.f, 0.f};

    // ---- dense part: rates @ Wab_T, K = 4096, 1024 per wave (32 tiles) ----
    {
        const bf16_t* arow = (const bf16_t*)ratesA + mrow * N_ + wave * 1024;
        const v16bf*  bp   = (const v16bf*)(wabP)
                           + ((long long)nt * KT_AB + wave * 32) * 32 + lane;
#pragma unroll 2
        for (int kt = 0; kt < 16; ++kt) {          // 16 pairs of k-tiles
            __builtin_prefetch((const void*)(bp + 64), 0, 1);
            v16bf a0 = load_a_frag(arow,      hi);
            v16bf a1 = load_a_frag(arow + 32, hi);
            v16bf b0 = bp[0];
            v16bf b1 = bp[32];
            acc0 = __builtin_amdgcn_wmma_f32_16x16x32_bf16(
                false, a0, false, b0, (short)0, acc0, false, false);
            acc1 = __builtin_amdgcn_wmma_f32_16x16x32_bf16(
                false, a1, false, b1, (short)0, acc1, false, false);
            arow += 64;
            bp   += 64;
        }
    }

    // ---- STP part (E columns only): Aux @ W_stp_T, K = 3072 --------------
    if (nt < NT_STP) {
        const bf16_t* arow = (const bf16_t*)auxA + mrow * NE_ + wave * 768;
        const v16bf*  bp   = (const v16bf*)(wstpP)
                           + ((long long)nt * KT_STP + wave * 24) * 32 + lane;
#pragma unroll 2
        for (int kt = 0; kt < 12; ++kt) {          // 12 pairs of k-tiles
            __builtin_prefetch((const void*)(bp + 64), 0, 1);
            v16bf a0 = load_a_frag(arow,      hi);
            v16bf a1 = load_a_frag(arow + 32, hi);
            v16bf b0 = bp[0];
            v16bf b1 = bp[32];
            acc0 = __builtin_amdgcn_wmma_f32_16x16x32_bf16(
                false, a0, false, b0, (short)0, acc0, false, false);
            acc1 = __builtin_amdgcn_wmma_f32_16x16x32_bf16(
                false, a1, false, b1, (short)0, acc1, false, false);
            arow += 64;
            bp   += 64;
        }
    }

#pragma unroll
    for (int r = 0; r < 8; ++r) acc0[r] += acc1[r];

    // ---- cross-wave reduction of the 16x16 f32 tile via LDS --------------
    __shared__ float red[4][32][8];
#pragma unroll
    for (int r = 0; r < 8; ++r) red[wave][lane][r] = acc0[r];
    __syncthreads();

    if (wave == 0) {
#pragma unroll
        for (int w = 1; w < 4; ++w)
#pragma unroll
            for (int r = 0; r < 8; ++r) acc0[r] += red[w][lane][r];

        // C/D layout: VGPR r, lane l -> (M = r + 8*(l>=16), N = l & 15)
        const int ncol = nt * 16 + (lane & 15);
        const int bb0  = hi * 8;
#pragma unroll
        for (int r = 0; r < 8; ++r) {
            int   b    = bb0 + r;
            int   idx  = b * N_ + ncol;
            long long fidx = ((long long)b * T_ + t) * N_ + ncol;
            float h    = acc0[r];
            float rc   = rec[idx] * E_SYN + h * (1.0f - E_SYN);
            float net  = ff[fidx] + rc;
            float nl   = fmaxf(net - thresh[idx], 0.0f);
            float rnew = rates[idx] * E_TAU + nl * (1.0f - E_TAU);
            rec[idx]         = rc;
            rates[idx]       = rnew;
            ratesB_next[idx] = __float2bfloat16(rnew);
            out[fidx]        = rnew;
        }
    }
}

// ---------------------------------------------------------------------------
extern "C" void kernel_launch(void* const* d_in, const int* in_sizes, int n_in,
                              void* d_out, int out_size, void* d_ws, size_t ws_size,
                              hipStream_t stream)
{
    const float* ff     = (const float*)d_in[0];   // [B,T,N]
    const float* Wab    = (const float*)d_in[1];   // [N,N]
    const float* Wstp   = (const float*)d_in[2];   // [NE,NE]
    const float* thresh = (const float*)d_in[3];   // [B,N]
    const float* rec0   = (const float*)d_in[4];   // [B,N]
    float*       out    = (float*)d_out;           // [B,T,N]

    char*  ws  = (char*)d_ws;
    size_t off = 0;
    auto alloc = [&](size_t bytes) -> void* {
        off = (off + 255) & ~(size_t)255;
        void* p = ws + off;
        off += bytes;
        return p;
    };

    __hip_bfloat16* wabP   = (__hip_bfloat16*)alloc((size_t)N_  * N_  * 2); // 32 MB
    __hip_bfloat16* wstpP  = (__hip_bfloat16*)alloc((size_t)NE_ * NE_ * 2); // 18 MB
    float* rates = (float*)alloc((size_t)B_ * N_  * 4);
    float* rec   = (float*)alloc((size_t)B_ * N_  * 4);
    float* u     = (float*)alloc((size_t)B_ * NE_ * 4);
    float* x     = (float*)alloc((size_t)B_ * NE_ * 4);
    __hip_bfloat16* ratesbf0 = (__hip_bfloat16*)alloc((size_t)B_ * N_  * 2);
    __hip_bfloat16* ratesbf1 = (__hip_bfloat16*)alloc((size_t)B_ * N_  * 2);
    __hip_bfloat16* auxbf    = (__hip_bfloat16*)alloc((size_t)B_ * NE_ * 2);
    (void)ws_size; (void)in_sizes; (void)n_in; (void)out_size;

    // one-time bf16 pack (weights then live in 192MB L2 across all steps)
    {
        long long tot = (long long)N_ * N_;
        pack_w_bf16<<<(unsigned)((tot + 255) / 256), 256, 0, stream>>>(Wab, wabP, N_, N_);
    }
    {
        long long tot = (long long)NE_ * NE_;
        pack_w_bf16<<<(unsigned)((tot + 255) / 256), 256, 0, stream>>>(Wstp, wstpP, NE_, NE_);
    }

    init_state<<<(B_ * N_) / 256, 256, 0, stream>>>(
        ff, rec0, thresh, rates, rec, u, x, ratesbf0);

    for (int t = 0; t < T_; ++t) {
        stp_step<<<(B_ * NE_) / 256, 256, 0, stream>>>(rates, u, x, auxbf);
        __hip_bfloat16* rA = (t & 1) ? ratesbf1 : ratesbf0;
        __hip_bfloat16* rB = (t & 1) ? ratesbf0 : ratesbf1;
        gemm_step<<<NT_AB, 128, 0, stream>>>(
            rA, auxbf, wabP, wstpP, ff, thresh, rates, rec, rB, out, t);
    }
}
